// MHA_54614804136500
// MI455X (gfx1250) — 8-bit matrix operand order repaired and checked by reading, compiled, not run on hardware
//
#include <hip/hip_runtime.h>
#include <stdint.h>


typedef int v8i __attribute__((ext_vector_type(8)));

#define B_DIM    8
#define N_DIM    1024
#define D_DIM    1024
#define HEADS    16
#define DH       64
#define INNER    1024
#define BN       (B_DIM * N_DIM)
#define BH       (B_DIM * HEADS)

#define SLOT_H    0
#define SLOT_W1   1
#define SLOT_W2   2
#define SLOT_G    3
#define SLOT_B    4
#define SLOT_QKV  5
#define SLOT_DOTS 6
#define SLOT_ATTN 7
#define SLOT_HOUT 8
#define SLOT_FIN  9

__device__ __forceinline__ float pot_scale(unsigned bits) {
    float m = __uint_as_float(bits);
    m = fmaxf(m, 1e-8f);
    return exp2f(ceilf(log2f(m * (1.0f / 127.0f))));
}
__device__ __forceinline__ int q_floor(float x, float inv_s) {
    return (int)fminf(fmaxf(floorf(x * inv_s), -128.0f), 127.0f);
}
__device__ __forceinline__ int q_round(float x, float inv_s) {
    return (int)fminf(fmaxf(rintf(x * inv_s), -128.0f), 127.0f);
}

__device__ __forceinline__ void async_copy16(unsigned ldsAddr, const void* gptr) {
    asm volatile("global_load_async_to_lds_b128 %0, %1, off"
                 :: "v"(ldsAddr), "v"((unsigned long long)(size_t)gptr)
                 : "memory");
}
__device__ __forceinline__ void async_wait0() {
    asm volatile("s_wait_asynccnt 0" ::: "memory");
}
__device__ __forceinline__ unsigned lds_off(const void* p) {
    return (unsigned)(size_t)p;
}

__global__ void init_scalars_kernel(unsigned* sc) {
    if (threadIdx.x < 16) sc[threadIdx.x] = 0u;
}

__global__ __launch_bounds__(256)
void absmax_kernel(const float* __restrict__ x, long long n, unsigned* __restrict__ out) {
    long long i = (long long)blockIdx.x * blockDim.x + threadIdx.x;
    long long stride = (long long)gridDim.x * blockDim.x;
    float lmax = 0.0f;
    for (; i < n; i += stride) lmax = fmaxf(lmax, fabsf(x[i]));
    for (int m = 16; m; m >>= 1) lmax = fmaxf(lmax, __shfl_xor(lmax, m, 32));
    if ((threadIdx.x & 31) == 0) atomicMax(out, __float_as_uint(lmax));
}

__global__ __launch_bounds__(256)
void ln_kernel(const float* __restrict__ x, const float* __restrict__ gamma,
               const float* __restrict__ beta, const unsigned* __restrict__ sc,
               float* __restrict__ h, unsigned* __restrict__ habs) {
    __shared__ float red[256];
    const int tid = threadIdx.x;
    const size_t row = blockIdx.x;
    const float* xr = x + row * D_DIM;
    float v[4];
    float sum = 0.0f;
#pragma unroll
    for (int i = 0; i < 4; ++i) { v[i] = xr[tid + i * 256]; sum += v[i]; }
    red[tid] = sum; __syncthreads();
    for (int s = 128; s; s >>= 1) { if (tid < s) red[tid] += red[tid + s]; __syncthreads(); }
    const float mu = red[0] * (1.0f / D_DIM);
    __syncthreads();
    float sq = 0.0f;
#pragma unroll
    for (int i = 0; i < 4; ++i) { float d = v[i] - mu; sq += d * d; }
    red[tid] = sq; __syncthreads();
    for (int s = 128; s; s >>= 1) { if (tid < s) red[tid] += red[tid + s]; __syncthreads(); }
    const float var = red[0] * (1.0f / D_DIM);
    const float rs = rsqrtf(var + 1e-5f);
    const float sg = pot_scale(sc[SLOT_G]);
    const float sb = pot_scale(sc[SLOT_B]);
    float lmax = 0.0f;
#pragma unroll
    for (int i = 0; i < 4; ++i) {
        const int d = tid + i * 256;
        const float gq = fminf(fmaxf(rintf(gamma[d] / sg), -128.0f), 127.0f) * sg;
        const float bq = fminf(fmaxf(rintf(beta[d]  / sb), -128.0f), 127.0f) * sb;
        const float hv = (v[i] - mu) * rs * gq + bq;
        h[row * D_DIM + d] = hv;
        lmax = fmaxf(lmax, fabsf(hv));
    }
    for (int m = 16; m; m >>= 1) lmax = fmaxf(lmax, __shfl_xor(lmax, m, 32));
    if ((tid & 31) == 0) atomicMax(habs, __float_as_uint(lmax));
}

__global__ __launch_bounds__(256)
void quant_kernel(const float* __restrict__ src, int8_t* __restrict__ dst,
                  long long n, const unsigned* __restrict__ slot) {
    long long i = (long long)blockIdx.x * blockDim.x + threadIdx.x;
    if (i >= n) return;
    const float inv_s = 1.0f / pot_scale(*slot);
    dst[i] = (int8_t)q_floor(src[i], inv_s);
}

__global__ __launch_bounds__(256)
void quant_wT_kernel(const float* __restrict__ src, int8_t* __restrict__ dstT,
                     int K, int N, const unsigned* __restrict__ slot) {
    const long long idx = (long long)blockIdx.x * blockDim.x + threadIdx.x;
    if (idx >= (long long)K * N) return;
    const float inv_s = 1.0f / pot_scale(*slot);
    const int n = (int)(idx / K), k = (int)(idx % K);
    dstT[idx] = (int8_t)q_round(src[(size_t)k * N + n], inv_s);
}

__global__ __launch_bounds__(256)
void gemm_i8_kernel(const int8_t* __restrict__ A, const int8_t* __restrict__ Bt,
                    float* __restrict__ C, int M, int N, int K,
                    int lda, int ldbt, int ldc,
                    long long strideA, long long strideB, long long strideC,
                    const unsigned* __restrict__ sAbits,
                    const unsigned* __restrict__ sBbits,
                    unsigned* __restrict__ absOut) {
    __shared__ __align__(16) int8_t lds_a[2 * 64 * 64];
    __shared__ __align__(16) int8_t lds_b[2 * 64 * 64];
    const int tid = threadIdx.x;
    const int lane = tid & 31, wave = tid >> 5;
    const int half = lane >> 4, l15 = lane & 15;
    const long long bz = blockIdx.z;
    const int mBase = blockIdx.y * 64;
    const int nBase = blockIdx.x * 64;
    A  += bz * strideA; Bt += bz * strideB; C += bz * strideC;
    const int mOff = (wave >> 1) * 16;
    const int nOff = (wave & 1) * 32;

    const int row4 = tid >> 2;
    const int c16  = (tid & 3) * 16;
    const unsigned ldsA = lds_off(&lds_a[row4 * 64 + c16]);
    const unsigned ldsB = lds_off(&lds_b[row4 * 64 + c16]);
    const int8_t* gA = A  + (size_t)(mBase + row4) * lda  + c16;
    const int8_t* gB = Bt + (size_t)(nBase + row4) * ldbt + c16;

    async_copy16(ldsA, gA);
    async_copy16(ldsB, gB);

    v8i acc0 = {0, 0, 0, 0, 0, 0, 0, 0};
    v8i acc1 = {0, 0, 0, 0, 0, 0, 0, 0};
    int buf = 0;
    for (int k0 = 0; k0 < K; k0 += 64) {
        async_wait0();
        __syncthreads();
        if (k0 + 64 < K) {
            const unsigned o = (unsigned)((buf ^ 1) << 12);
            async_copy16(ldsA + o, gA + k0 + 64);
            async_copy16(ldsB + o, gB + k0 + 64);
        }
        const int8_t* la = lds_a + (buf << 12);
        const int8_t* lb = lds_b + (buf << 12);
        v8i af, bf0, bf1;
#pragma unroll
        for (int i = 0; i < 8; ++i)
            af[i] = *(const int*)(&la[(mOff + l15) * 64 + (i >> 1) * 16 + half * 8 + (i & 1) * 4]);
#pragma unroll
        for (int i = 0; i < 8; ++i) {
            const int ko = (i >> 1) * 16 + half * 8 + (i & 1) * 4;
            bf0[i] = *(const int*)(&lb[(nOff      + l15) * 64 + ko]);
            bf1[i] = *(const int*)(&lb[(nOff + 16 + l15) * 64 + ko]);
        }
        acc0 = __builtin_amdgcn_wmma_i32_16x16x64_iu8(true, af, true, bf0, acc0, false, false);
        acc1 = __builtin_amdgcn_wmma_i32_16x16x64_iu8(true, af, true, bf1, acc1, false, false);
        __syncthreads();
        buf ^= 1;
    }
    const float s = pot_scale(*sAbits) * pot_scale(*sBbits);
    float lmax = 0.0f;
#pragma unroll
    for (int nn = 0; nn < 2; ++nn) {
        const v8i acc = nn ? acc1 : acc0;
        const int col = nBase + nOff + nn * 16 + l15;
#pragma unroll
        for (int i = 0; i < 8; ++i) {
            const int row = mBase + mOff + i + half * 8;
            const float v = (float)acc[i] * s;
            C[(size_t)row * ldc + col] = v;
            lmax = fmaxf(lmax, fabsf(v));
        }
    }
    for (int m = 16; m; m >>= 1) lmax = fmaxf(lmax, __shfl_xor(lmax, m, 32));
    if (lane == 0) atomicMax(absOut, __float_as_uint(lmax));
}

__global__ __launch_bounds__(256)
void qkv_split_kernel(const float* __restrict__ qkv, int8_t* __restrict__ qi8,
                      const unsigned* __restrict__ slot) {
    const long long idx = (long long)blockIdx.x * blockDim.x + threadIdx.x;
    const long long total = 3LL * BH * N_DIM * DH;
    if (idx >= total) return;
    const float inv_s = 1.0f / pot_scale(*slot);
    const int which = (int)(idx / 8388608LL);
    const long long rem = idx % 8388608LL;
    const int bh = (int)(rem / 65536);
    const int n  = (int)((rem >> 6) & 1023);
    const int d  = (int)(rem & 63);
    const int b  = bh >> 4, hh = bh & 15;
    const float v = qkv[(size_t)(b * N_DIM + n) * (3 * INNER) + which * INNER + hh * DH + d];
    long long dst = idx;
    if (which == 2)
        dst = 2LL * 8388608LL + (long long)bh * 65536 + (long long)d * N_DIM + n;
    qi8[dst] = (int8_t)q_floor(v, inv_s);
}

__global__ __launch_bounds__(256)
void dots_absmax_kernel(const int8_t* __restrict__ q, const int8_t* __restrict__ k,
                        const unsigned* __restrict__ sc, const float* __restrict__ log_temp,
                        unsigned* __restrict__ absOut) {
    const int bh = blockIdx.z, nt = blockIdx.x;
    const int wave = threadIdx.x >> 5, lane = threadIdx.x & 31;
    const int half = lane >> 4, l15 = lane & 15;
    const int mt = blockIdx.y * 8 + wave;
    const int8_t* qb = q + (size_t)bh * (N_DIM * DH);
    const int8_t* kb = k + (size_t)bh * (N_DIM * DH);
    v8i af, bf, acc = {0, 0, 0, 0, 0, 0, 0, 0};
#pragma unroll
    for (int i = 0; i < 8; ++i)
        af[i] = *(const int*)(qb + (size_t)(nt * 16 + l15) * DH + (i >> 1) * 16 + half * 8 + (i & 1) * 4);
#pragma unroll
    for (int i = 0; i < 8; ++i)
        bf[i] = *(const int*)(kb + (size_t)(mt * 16 + l15) * DH + (i >> 1) * 16 + half * 8 + (i & 1) * 4);
    acc = __builtin_amdgcn_wmma_i32_16x16x64_iu8(true, af, true, bf, acc, false, false);
    const float sq = pot_scale(sc[SLOT_QKV]);
    float t = expf(log_temp[0]); t = floorf(t * 4096.0f) * (1.0f / 4096.0f);
    const float dsc = sq * sq * t;
    float lmax = 0.0f;
#pragma unroll
    for (int i = 0; i < 8; ++i) lmax = fmaxf(lmax, fabsf((float)acc[i] * dsc));
    for (int m = 16; m; m >>= 1) lmax = fmaxf(lmax, __shfl_xor(lmax, m, 32));
    if (lane == 0) atomicMax(absOut, __float_as_uint(lmax));
}

__global__ __launch_bounds__(256)
void attn_stats_kernel(const int8_t* __restrict__ q, const int8_t* __restrict__ k,
                       int8_t* __restrict__ dots, const unsigned* __restrict__ sc,
                       const float* __restrict__ log_temp,
                       float* __restrict__ rowmax, float* __restrict__ rowsum,
                       unsigned* __restrict__ attnAbs) {
    __shared__ int sh_rmax[16];
    __shared__ float sh_sum[16];
    __shared__ int8_t sh_dq[16 * 1024];
    const int tid = threadIdx.x;
    const int bh = blockIdx.y, nt = blockIdx.x;
    const int wave = tid >> 5, lane = tid & 31;
    const int half = lane >> 4, l15 = lane & 15;
    if (tid < 16) { sh_rmax[tid] = -(1 << 30); sh_sum[tid] = 0.0f; }
    __syncthreads();
    const int8_t* qb = q + (size_t)bh * (N_DIM * DH);
    const int8_t* kb = k + (size_t)bh * (N_DIM * DH);
    const float sq = pot_scale(sc[SLOT_QKV]);
    float t = expf(log_temp[0]); t = floorf(t * 4096.0f) * (1.0f / 4096.0f);
    const float dsc = sq * sq * t;
    const float sd = pot_scale(sc[SLOT_DOTS]);
    const float inv_sd = 1.0f / sd;
    v8i af;
#pragma unroll
    for (int i = 0; i < 8; ++i)
        af[i] = *(const int*)(qb + (size_t)(nt * 16 + l15) * DH + (i >> 1) * 16 + half * 8 + (i & 1) * 4);
    int8_t* drow = dots + (size_t)bh * (N_DIM * N_DIM) + (size_t)nt * 16 * N_DIM;
    int rmax_reg[8];
#pragma unroll
    for (int i = 0; i < 8; ++i) rmax_reg[i] = -(1 << 30);
    for (int j = 0; j < 8; ++j) {
        const int mt = wave * 8 + j;
        v8i bf, acc = {0, 0, 0, 0, 0, 0, 0, 0};
#pragma unroll
        for (int i = 0; i < 8; ++i)
            bf[i] = *(const int*)(kb + (size_t)(mt * 16 + l15) * DH + (i >> 1) * 16 + half * 8 + (i & 1) * 4);
        acc = __builtin_amdgcn_wmma_i32_16x16x64_iu8(true, af, true, bf, acc, false, false);
#pragma unroll
        for (int i = 0; i < 8; ++i) {
            const int qv = (int)fminf(fmaxf(floorf((float)acc[i] * dsc * inv_sd), -128.0f), 127.0f);
            const int r = i + half * 8;
            const int m = mt * 16 + l15;
            sh_dq[r * 1024 + m] = (int8_t)qv;
            drow[(size_t)r * N_DIM + m] = (int8_t)qv;
            rmax_reg[i] = max(rmax_reg[i], qv);
        }
    }
#pragma unroll
    for (int i = 0; i < 8; ++i)
        for (int m = 8; m; m >>= 1) rmax_reg[i] = max(rmax_reg[i], __shfl_xor(rmax_reg[i], m, 32));
    if (l15 == 0)
#pragma unroll
        for (int i = 0; i < 8; ++i) atomicMax(&sh_rmax[i + half * 8], rmax_reg[i]);
    __syncthreads();
    {
        const int r = tid & 15, seg = tid >> 4;
        const float rm = (float)sh_rmax[r];
        float ssum = 0.0f;
        for (int m = seg * 64; m < seg * 64 + 64; ++m)
            ssum += expf(((float)sh_dq[r * 1024 + m] - rm) * sd);
        atomicAdd(&sh_sum[r], ssum);
    }
    __syncthreads();
    if (tid < 16) {
        const size_t rid = (size_t)bh * N_DIM + nt * 16 + tid;
        rowmax[rid] = (float)sh_rmax[tid] * sd;
        rowsum[rid] = sh_sum[tid];
        atomicMax(attnAbs, __float_as_uint(1.0f / sh_sum[tid]));
    }
}

__global__ __launch_bounds__(256)
void attn_quant_kernel(int8_t* __restrict__ dots, const float* __restrict__ rowmax,
                       const float* __restrict__ rowsum, const unsigned* __restrict__ sc) {
    const long long i4 = (long long)blockIdx.x * blockDim.x + threadIdx.x;
    const long long total4 = ((long long)BH * N_DIM * N_DIM) >> 2;
    if (i4 >= total4) return;
    const float sd = pot_scale(sc[SLOT_DOTS]);
    const float inv_sa = 1.0f / pot_scale(sc[SLOT_ATTN]);
    const long long row = (i4 << 2) >> 10;
    const float rm = rowmax[row];
    const float inv_rs = 1.0f / rowsum[row];
    int w = ((int*)dots)[i4];
    int o = 0;
#pragma unroll
    for (int j = 0; j < 4; ++j) {
        const int8_t d = (int8_t)(w >> (8 * j));
        const float a = expf((float)d * sd - rm) * inv_rs;
        o |= (q_floor(a, inv_sa) & 0xFF) << (8 * j);
    }
    ((int*)dots)[i4] = o;
}

__global__ __launch_bounds__(256)
void merge_quant_kernel(const float* __restrict__ headout, int8_t* __restrict__ oi8,
                        const unsigned* __restrict__ slot) {
    const long long idx = (long long)blockIdx.x * blockDim.x + threadIdx.x;
    if (idx >= (long long)BN * INNER) return;
    const float inv_s = 1.0f / pot_scale(*slot);
    const int r = (int)(idx >> 10), c = (int)(idx & 1023);
    const int b = r >> 10, n = r & 1023;
    const int hh = c >> 6, d = c & 63;
    const float v = headout[((size_t)(b * HEADS + hh) * N_DIM + n) * DH + d];
    oi8[idx] = (int8_t)q_floor(v, inv_s);
}

__global__ __launch_bounds__(256)
void final_quant_kernel(const float* __restrict__ src, float* __restrict__ dst,
                        long long n, const unsigned* __restrict__ slot) {
    const long long i = (long long)blockIdx.x * blockDim.x + threadIdx.x;
    if (i >= n) return;
    const float s = pot_scale(*slot);
    dst[i] = (float)q_floor(src[i], 1.0f / s) * s;
}

extern "C" void kernel_launch(void* const* d_in, const int* in_sizes, int n_in,
                              void* d_out, int out_size, void* d_ws, size_t ws_size,
                              hipStream_t stream) {
    (void)in_sizes; (void)n_in; (void)out_size; (void)ws_size;
    const float* x        = (const float*)d_in[0];
    const float* ln_gamma = (const float*)d_in[1];
    const float* ln_beta  = (const float*)d_in[2];
    const float* w_qkv    = (const float*)d_in[3];
    const float* w_out    = (const float*)d_in[4];
    const float* log_temp = (const float*)d_in[5];
    float* out = (float*)d_out;

    char* ws = (char*)d_ws;
    unsigned* sc     = (unsigned*)(ws + 0);
    float*   h_f32   = (float*)  (ws + 1024);
    int8_t*  h_i8    = (int8_t*) (ws + 1024 + 33554432ULL);
    int8_t*  wq_i8T  = (int8_t*) (ws + 1024 + 41943040ULL);
    int8_t*  wo_i8T  = (int8_t*) (ws + 1024 + 45088768ULL);
    float*   qkv_f32 = (float*)  (ws + 1024 + 46137344ULL);
    int8_t*  q_i8    = (int8_t*) (ws + 1024 + 146800640ULL);
    int8_t*  k_i8    = q_i8 + 8388608ULL;
    int8_t*  vT_i8   = k_i8 + 8388608ULL;
    int8_t*  dots_i8 = (int8_t*) (ws + 1024 + 171966464ULL);
    float*   rowmax  = (float*)  (ws + 1024 + 306184192ULL);
    float*   rowsum  = rowmax + (size_t)BH * N_DIM;
    float*   headout = h_f32;
    int8_t*  out_i8  = h_i8;
    float*   tmp2    = qkv_f32;

    init_scalars_kernel<<<1, 64, 0, stream>>>(sc);

    absmax_kernel<<<1024, 256, 0, stream>>>(w_qkv, (long long)D_DIM * 3 * INNER, sc + SLOT_W1);
    absmax_kernel<<<512,  256, 0, stream>>>(w_out, (long long)INNER * D_DIM,     sc + SLOT_W2);
    absmax_kernel<<<4,    256, 0, stream>>>(ln_gamma, D_DIM, sc + SLOT_G);
    absmax_kernel<<<4,    256, 0, stream>>>(ln_beta,  D_DIM, sc + SLOT_B);

    ln_kernel<<<BN, 256, 0, stream>>>(x, ln_gamma, ln_beta, sc, h_f32, sc + SLOT_H);

    quant_kernel<<<(BN * D_DIM) / 256, 256, 0, stream>>>(h_f32, h_i8, (long long)BN * D_DIM, sc + SLOT_H);
    quant_wT_kernel<<<(D_DIM * 3 * INNER) / 256, 256, 0, stream>>>(w_qkv, wq_i8T, D_DIM, 3 * INNER, sc + SLOT_W1);
    quant_wT_kernel<<<(INNER * D_DIM) / 256, 256, 0, stream>>>(w_out, wo_i8T, INNER, D_DIM, sc + SLOT_W2);

    gemm_i8_kernel<<<dim3(3 * INNER / 64, BN / 64, 1), 256, 0, stream>>>(
        h_i8, wq_i8T, qkv_f32, BN, 3 * INNER, D_DIM, D_DIM, D_DIM, 3 * INNER,
        0, 0, 0, sc + SLOT_H, sc + SLOT_W1, sc + SLOT_QKV);

    qkv_split_kernel<<<(3 * BH * N_DIM * DH) / 256, 256, 0, stream>>>(qkv_f32, q_i8, sc + SLOT_QKV);

    dots_absmax_kernel<<<dim3(N_DIM / 16, 8, BH), 256, 0, stream>>>(q_i8, k_i8, sc, log_temp, sc + SLOT_DOTS);

    attn_stats_kernel<<<dim3(N_DIM / 16, BH), 256, 0, stream>>>(
        q_i8, k_i8, dots_i8, sc, log_temp, rowmax, rowsum, sc + SLOT_ATTN);

    attn_quant_kernel<<<(int)(((long long)BH * N_DIM * N_DIM) / 1024), 256, 0, stream>>>(
        dots_i8, rowmax, rowsum, sc);

    gemm_i8_kernel<<<dim3(DH / 64, N_DIM / 64, BH), 256, 0, stream>>>(
        dots_i8, vT_i8, headout, N_DIM, DH, N_DIM, N_DIM, N_DIM, DH,
        (long long)N_DIM * N_DIM, (long long)N_DIM * DH, (long long)N_DIM * DH,
        sc + SLOT_ATTN, sc + SLOT_QKV, sc + SLOT_HOUT);

    merge_quant_kernel<<<(BN * INNER) / 256, 256, 0, stream>>>(headout, out_i8, sc + SLOT_HOUT);

    gemm_i8_kernel<<<dim3(D_DIM / 64, BN / 64, 1), 256, 0, stream>>>(
        out_i8, wo_i8T, tmp2, BN, D_DIM, INNER, INNER, INNER, D_DIM,
        0, 0, 0, sc + SLOT_HOUT, sc + SLOT_W2, sc + SLOT_FIN);

    final_quant_kernel<<<(BN * D_DIM) / 256, 256, 0, stream>>>(tmp2, out, (long long)BN * D_DIM, sc + SLOT_FIN);
}
